// MyRNN_41394894799448
// MI455X (gfx1250) — matrix operand order repaired and checked by reading, not compiled, not run on hardware
//
#include <hip/hip_runtime.h>
#include <hip/hip_bf16.h>

// ---------------------------------------------------------------------------
// 2-layer GRU (reset_after) over T=80 steps, B=128, U=1024, EMB=100 (pad 128)
// bf16 WMMA (v_wmma_f32_16x16x32_bf16) with f32 accumulation + fused gates.
// h kept as f32 (recurrence math) + bf16 mirror (WMMA A fragments, no cvt).
// ---------------------------------------------------------------------------

typedef __attribute__((ext_vector_type(16))) __bf16 v16bf;
typedef __attribute__((ext_vector_type(8)))  float  v8f;

#define GRU_UNITS   1024
#define GRU_3U      3072
#define GRU_BATCH   128
#define GRU_SEQ     80
#define GRU_EMB     100
#define GRU_EMB_PAD 128
#define GRU_VOCAB   10000

// ---- WMMA helpers ---------------------------------------------------------

__device__ __forceinline__ v8f wmma_bf16(v16bf a, v16bf b, v8f c) {
  // 8 args: (neg_a, A, neg_b, B, c_mod, C, reuse_a, reuse_b)
  return __builtin_amdgcn_wmma_f32_16x16x32_bf16(false, a, false, b,
                                                 (short)0, c, false, false);
}

// A-fragment (16x32, MxK): lane L holds row M=L%16, K = (L/16)*8 + {0..7} and
// (L/16)*8 + {16..23}.  Source is a row-major bf16 matrix with leading dim ld.
__device__ __forceinline__ v16bf a_frag_bf16(const __bf16* __restrict__ base,
                                             int ld, int row, int k0, int half) {
  const __bf16* p0 = base + (size_t)row * ld + k0 + half * 8;
  v16bf a;
#pragma unroll
  for (int i = 0; i < 8; ++i) {
    a[i]     = p0[i];
    a[8 + i] = p0[16 + i];
  }
  return a;
}

// B-fragment (32x16, KxN): lane L holds col N = j + L%16, K = (L/16)*8 + {0..7} and
// (L/16)*8 + {16..23} (k passed as k0 + L).
__device__ __forceinline__ v16bf b_frag(const __bf16* __restrict__ W,
                                        int k, int j) {
  const int kb = k & ~31, l = k & 31;
  const __bf16* p0 = W + (size_t)(kb + (l >> 4) * 8) * GRU_3U + j + (l & 15);
  v16bf b;
#pragma unroll
  for (int i = 0; i < 8; ++i) {
    b[i]     = p0[(size_t)i * GRU_3U];
    b[8 + i] = p0[(size_t)(16 + i) * GRU_3U];
  }
  return b;
}

__device__ __forceinline__ float sigmoidf_dev(float x) {
  return 1.0f / (1.0f + __expf(-x));
}

// ---- One GRU timestep for one layer ---------------------------------------
// Output: hNew[128,1024] (f32) + hNewB (bf16 mirror for next step's A frags).
// Grid: 64 blocks x 256 threads = 512 waves.
// Wave -> (row-tile mt in 0..7, col-tile nt in 0..63); 6 accumulators
// (z/r/n for gx and gh) fused with the gate nonlinearity + h update.
template <bool LAYER1>
__global__ __launch_bounds__(256)
void gru_step_kernel(const __bf16* __restrict__ xB,       // LAYER2: input x bf16 (=h0 new) [128,1024]
                     const __bf16* __restrict__ Epad,     // LAYER1: padded bf16 embedding [VOCAB,128]
                     const int* __restrict__ tokens, int t,
                     const __bf16* __restrict__ WxB,      // [KX, 3072] bf16 (KX=128 or 1024)
                     const __bf16* __restrict__ WhB,      // [1024, 3072] bf16
                     const float* __restrict__ bx,
                     const float* __restrict__ bh,
                     const float* __restrict__ hOld,      // [128,1024] f32
                     const __bf16* __restrict__ hOldB,    // [128,1024] bf16 mirror
                     float* __restrict__ hNew,            // [128,1024] f32
                     __bf16* __restrict__ hNewB)          // [128,1024] bf16 mirror
{
  const int lane    = threadIdx.x & 31;
  const int wid     = (int)((blockIdx.x * blockDim.x + threadIdx.x) >> 5);
  const int mt      = wid >> 6;                 // 0..7  row tile
  const int nt      = wid & 63;                 // 0..63 col tile
  const int rowBase = mt * 16;
  const int j0      = nt * 16;
  const int half    = lane >> 4;
  const int mrow    = rowBase + (lane & 15);    // A-fragment row for this lane

  v8f axz = {}, axr = {}, axn = {};             // gx accumulators (z,r,n)
  v8f ahz = {}, ahr = {}, ahn = {};             // gh accumulators (z,r,n)

  // ---- gx = x @ Wx ----
  if (LAYER1) {
    const int tok = tokens[mrow * GRU_SEQ + t];
    const __bf16* xrow = Epad + (size_t)tok * GRU_EMB_PAD;
#pragma unroll
    for (int k0 = 0; k0 < GRU_EMB_PAD; k0 += 32) {
      v16bf a;
      const __bf16* p0 = xrow + k0 + half * 8;
#pragma unroll
      for (int i = 0; i < 8; ++i) { a[i] = p0[i]; a[8 + i] = p0[16 + i]; }
      v16bf b0 = b_frag(WxB, k0 + lane, j0);
      v16bf b1 = b_frag(WxB, k0 + lane, j0 + GRU_UNITS);
      v16bf b2 = b_frag(WxB, k0 + lane, j0 + 2 * GRU_UNITS);
      axz = wmma_bf16(a, b0, axz);
      axr = wmma_bf16(a, b1, axr);
      axn = wmma_bf16(a, b2, axn);
    }
  } else {
    for (int k0 = 0; k0 < GRU_UNITS; k0 += 32) {
      v16bf a  = a_frag_bf16(xB, GRU_UNITS, mrow, k0, half);
      v16bf b0 = b_frag(WxB, k0 + lane, j0);
      v16bf b1 = b_frag(WxB, k0 + lane, j0 + GRU_UNITS);
      v16bf b2 = b_frag(WxB, k0 + lane, j0 + 2 * GRU_UNITS);
      axz = wmma_bf16(a, b0, axz);
      axr = wmma_bf16(a, b1, axr);
      axn = wmma_bf16(a, b2, axn);
    }
  }

  // ---- gh = h @ Wh ----
  for (int k0 = 0; k0 < GRU_UNITS; k0 += 32) {
    if (k0 + 32 < GRU_UNITS) {  // pull next B tile toward the WGP (global_prefetch)
      __builtin_prefetch((const void*)(WhB + (size_t)(k0 + 32 + lane) * GRU_3U + j0), 0, 1);
    }
    v16bf a  = a_frag_bf16(hOldB, GRU_UNITS, mrow, k0, half);
    v16bf b0 = b_frag(WhB, k0 + lane, j0);
    v16bf b1 = b_frag(WhB, k0 + lane, j0 + GRU_UNITS);
    v16bf b2 = b_frag(WhB, k0 + lane, j0 + 2 * GRU_UNITS);
    ahz = wmma_bf16(a, b0, ahz);
    ahr = wmma_bf16(a, b1, ahr);
    ahn = wmma_bf16(a, b2, ahn);
  }

  // ---- fused gates + state update on the C layout:
  // (VGPR g, lane L) -> M = g + 8*(L/16), N = L%16
  const int nn = lane & 15;
  const int u  = j0 + nn;
  const float bxz = bx[u], bxr = bx[GRU_UNITS + u], bxn = bx[2 * GRU_UNITS + u];
  const float bhz = bh[u], bhr = bh[GRU_UNITS + u], bhn = bh[2 * GRU_UNITS + u];
#pragma unroll
  for (int g = 0; g < 8; ++g) {
    const int row = rowBase + g + half * 8;
    const float xz = axz[g] + bxz;
    const float xr = axr[g] + bxr;
    const float xn = axn[g] + bxn;
    const float gz = ahz[g] + bhz;
    const float gr = ahr[g] + bhr;
    const float gn = ahn[g] + bhn;
    const float z  = sigmoidf_dev(xz + gz);
    const float r  = sigmoidf_dev(xr + gr);
    const float nv = tanhf(xn + r * gn);       // reset_after: r AFTER recurrent matmul
    const float ho = hOld[(size_t)row * GRU_UNITS + u];
    const float hn_val = z * ho + (1.0f - z) * nv;
    hNew [(size_t)row * GRU_UNITS + u] = hn_val;
    hNewB[(size_t)row * GRU_UNITS + u] = (__bf16)hn_val;
  }
}

// ---- conversion / padding: dst[r*dcols+c] = (r<srows && c<scols) ? src : 0 -
__global__ void pad_convert_kernel(const float* __restrict__ src,
                                   __bf16* __restrict__ dst,
                                   int srows, int scols, int dcols, int total) {
  int i = blockIdx.x * blockDim.x + threadIdx.x;
  if (i >= total) return;
  int r = i / dcols, c = i - r * dcols;
  float v = (r < srows && c < scols) ? src[(size_t)r * scols + c] : 0.0f;
  dst[i] = (__bf16)v;
}

__global__ void zero_b32_kernel(unsigned int* __restrict__ p, int n) {
  int i = blockIdx.x * blockDim.x + threadIdx.x;
  if (i < n) p[i] = 0u;
}

// ---- final dense -> sigmoid ------------------------------------------------
__global__ void final_dense_kernel(const float* __restrict__ h1,
                                   const float* __restrict__ Wf,
                                   const float* __restrict__ bfin,
                                   float* __restrict__ out) {
  int b = blockIdx.x * blockDim.x + threadIdx.x;
  if (b >= GRU_BATCH) return;
  float s = bfin[0];
  for (int u = 0; u < GRU_UNITS; ++u) s += h1[(size_t)b * GRU_UNITS + u] * Wf[u];
  out[b] = 1.0f / (1.0f + __expf(-s));
}

// ---------------------------------------------------------------------------

extern "C" void kernel_launch(void* const* d_in, const int* in_sizes, int n_in,
                              void* d_out, int out_size, void* d_ws, size_t ws_size,
                              hipStream_t stream) {
  const int*   tokens = (const int*)  d_in[0];
  const float* E      = (const float*)d_in[1];
  const float* Wx1    = (const float*)d_in[2];
  const float* Wh1    = (const float*)d_in[3];
  const float* bx1    = (const float*)d_in[4];
  const float* bh1    = (const float*)d_in[5];
  const float* Wx2    = (const float*)d_in[6];
  const float* Wh2    = (const float*)d_in[7];
  const float* bx2    = (const float*)d_in[8];
  const float* bh2    = (const float*)d_in[9];
  const float* Wf     = (const float*)d_in[10];
  const float* bf_    = (const float*)d_in[11];
  float* out = (float*)d_out;

  // ---- workspace layout (256B aligned) ----
  char* ws = (char*)d_ws;
  size_t off = 0;
  auto take = [&](size_t bytes) -> char* {
    char* p = ws + off;
    off = (off + bytes + 255) & ~(size_t)255;
    return p;
  };
  const size_t WHB_BYTES = (size_t)GRU_UNITS * GRU_3U * sizeof(__bf16);    // 6 MB
  __bf16* Wh1b = (__bf16*)take(WHB_BYTES);
  __bf16* Wx2b = (__bf16*)take(WHB_BYTES);
  __bf16* Wh2b = (__bf16*)take(WHB_BYTES);
  __bf16* Wx1p = (__bf16*)take((size_t)GRU_EMB_PAD * GRU_3U * sizeof(__bf16));
  __bf16* Epad = (__bf16*)take((size_t)GRU_VOCAB * GRU_EMB_PAD * sizeof(__bf16));
  const size_t H_BYTES  = (size_t)GRU_BATCH * GRU_UNITS * sizeof(float);   // 512 KB
  const size_t HB_BYTES = (size_t)GRU_BATCH * GRU_UNITS * sizeof(__bf16);  // 256 KB
  // f32 state (4 contiguous buffers), then bf16 mirrors (4 contiguous)
  float*  h0a  = (float*) take(H_BYTES);
  float*  h0b  = (float*) take(H_BYTES);
  float*  h1a  = (float*) take(H_BYTES);
  float*  h1b  = (float*) take(H_BYTES);
  __bf16* h0aB = (__bf16*)take(HB_BYTES);
  __bf16* h0bB = (__bf16*)take(HB_BYTES);
  __bf16* h1aB = (__bf16*)take(HB_BYTES);
  __bf16* h1bB = (__bf16*)take(HB_BYTES);
  (void)ws_size; (void)in_sizes; (void)n_in; (void)out_size;

  // ---- one-time (per launch) conversions, deterministic ----
  {
    const int tWh = GRU_UNITS * GRU_3U;                 // 3,145,728
    pad_convert_kernel<<<(tWh + 255) / 256, 256, 0, stream>>>(Wh1, Wh1b, GRU_UNITS, GRU_3U, GRU_3U, tWh);
    pad_convert_kernel<<<(tWh + 255) / 256, 256, 0, stream>>>(Wx2, Wx2b, GRU_UNITS, GRU_3U, GRU_3U, tWh);
    pad_convert_kernel<<<(tWh + 255) / 256, 256, 0, stream>>>(Wh2, Wh2b, GRU_UNITS, GRU_3U, GRU_3U, tWh);
    const int tWx1 = GRU_EMB_PAD * GRU_3U;              // pad 100 -> 128 K-rows with zeros
    pad_convert_kernel<<<(tWx1 + 255) / 256, 256, 0, stream>>>(Wx1, Wx1p, GRU_EMB, GRU_3U, GRU_3U, tWx1);
    const int tE = GRU_VOCAB * GRU_EMB_PAD;             // pad 100 -> 128 cols with zeros
    pad_convert_kernel<<<(tE + 255) / 256, 256, 0, stream>>>(E, Epad, GRU_VOCAB, GRU_EMB, GRU_EMB_PAD, tE);
    // zero all h state: 4 f32 buffers + 4 bf16 mirrors (contiguous, dword-sized)
    const int tZ = (int)((4 * H_BYTES + 4 * HB_BYTES) / 4);
    zero_b32_kernel<<<(tZ + 255) / 256, 256, 0, stream>>>((unsigned int*)h0a, tZ);
  }

  // ---- the scan: 80 timesteps, ping-pong h in workspace ----
  float*  H0 [2] = {h0a,  h0b};
  float*  H1 [2] = {h1a,  h1b};
  __bf16* H0B[2] = {h0aB, h0bB};
  __bf16* H1B[2] = {h1aB, h1bB};
  for (int t = 0; t < GRU_SEQ; ++t) {
    const int rd = t & 1;
    const int wr = rd ^ 1;
    // layer 1: x from embedding, h = H0
    gru_step_kernel<true><<<64, 256, 0, stream>>>(
        nullptr, Epad, tokens, t, Wx1p, Wh1b, bx1, bh1,
        H0[rd], H0B[rd], H0[wr], H0B[wr]);
    // layer 2: x = new H0 (bf16 mirror), h = H1
    gru_step_kernel<false><<<64, 256, 0, stream>>>(
        H0B[wr], nullptr, nullptr, 0, Wx2b, Wh2b, bx2, bh2,
        H1[rd], H1B[rd], H1[wr], H1B[wr]);
  }

  // after t = 79 (rd=1, wr=0), final hidden state is H1[0]
  final_dense_kernel<<<1, 128, 0, stream>>>(H1[0], Wf, bf_, out);
}
